// SchemaEncoder_32641751450078
// MI455X (gfx1250) — matrix operand order repaired and checked by reading, not compiled, not run on hardware
//
#include <hip/hip_runtime.h>
#include <hip/hip_fp16.h>

#define HIDDEN 512
#define LOWER 128
#define NUM_RELS 6

typedef __attribute__((ext_vector_type(16))) _Float16 v16h;
typedef __attribute__((ext_vector_type(8)))  _Float16 v8h;
typedef __attribute__((ext_vector_type(4)))  _Float16 v4h;
typedef __attribute__((ext_vector_type(8)))  float    v8f;
typedef __attribute__((ext_vector_type(4)))  float    v4f;

union V16H { v16h v; v8h h[2]; };

// ---------------------------------------------------------------------------
// Fragment loaders (wave32), contiguous k order k = k0 + 16*(lane/16) + s for A and B
// A: 16x32 f16 tile, row-major source with row stride K halves.
//   lane L (0-15): row M=L, holds K = k0+0..15 (VGPR0-7)
//   lane L+16:     row M=L, holds K = k0+16..31
__device__ __forceinline__ v16h load_a_frag(const _Float16* __restrict__ A,
                                            int K, int row0, int k0, int M) {
  const int lane = threadIdx.x & 31;
  int r = row0 + (lane & 15);
  if (r >= M) r = M - 1;                    // clamp (stores are guarded)
  const int ks = (lane >> 4) << 4;          // 0 or 16
  const _Float16* p = A + (size_t)r * K + (k0 + ks);
  V16H u;
  u.h[0] = *(const v8h*)(p);                // 16B contiguous
  u.h[1] = *(const v8h*)(p + 8);            // 16B contiguous
  return u.v;
}

// B: 32x16 f16 tile from pre-transposed weights WT[N][K] (row-major, stride K).
//   lane L (0-15): col N=L,    K = k0+0..15  (one contiguous 32B chunk)
//   lane L+16:     col N=L-16, K = k0+16..31
__device__ __forceinline__ v16h load_b_frag(const _Float16* __restrict__ WT,
                                            int K, int col0, int k0) {
  const int lane = threadIdx.x & 31;
  const int n = col0 + (lane & 15);
  const int kh = (lane >> 4) << 4;          // 0 or 16
  return *(const v16h*)(WT + (size_t)n * K + (k0 + kh));
}

// ---------------------------------------------------------------------------
// Generic WMMA GEMM: out[z] = act(A @ WT[z]^T + bias)
// Block: 256 threads = 8 waves (4 along M, 2 along N). Wave tile 16x64.
// Block tile 64(M) x 128(N). Requires Nout % 128 == 0, K % 32 == 0.
template <bool RELU, bool F16OUT, bool HASBIAS>
__global__ __launch_bounds__(256) void wmma_gemm_kernel(
    const _Float16* __restrict__ A, const _Float16* __restrict__ WT,
    const float* __restrict__ bias, void* __restrict__ out,
    int M, int K, int Nout, long wtBatchStride, long outBatchStride) {
  const int z = blockIdx.z;
  WT += (size_t)z * wtBatchStride;
  _Float16* out16 = (_Float16*)out + (size_t)z * outBatchStride;
  float*    out32 = (float*)out    + (size_t)z * outBatchStride;

  const int wave  = threadIdx.x >> 5;
  const int row0  = blockIdx.y * 64  + (wave & 3) * 16;
  const int col0  = blockIdx.x * 128 + (wave >> 2) * 64;

  v8f acc[4] = {};
  for (int k0 = 0; k0 < K; k0 += 32) {
    v16h a = load_a_frag(A, K, row0, k0, M);
#pragma unroll
    for (int t = 0; t < 4; ++t) {
      v16h b = load_b_frag(WT, K, col0 + t * 16, k0);
      acc[t] = __builtin_amdgcn_wmma_f32_16x16x32_f16(
          false, a, false, b, (short)0, acc[t], false, false);
    }
  }

  // C/D layout: lane&15 = col, (lane>>4)*8 + v = row
  const int lane = threadIdx.x & 31;
  const int n  = lane & 15;
  const int mo = (lane >> 4) << 3;
#pragma unroll
  for (int t = 0; t < 4; ++t) {
    const int col = col0 + t * 16 + n;
    const float bv = HASBIAS ? bias[col] : 0.0f;
#pragma unroll
    for (int v = 0; v < 8; ++v) {
      const int r = row0 + mo + v;
      if (r < M) {
        float x = acc[t][v] + bv;
        if (RELU) x = x > 0.0f ? x : 0.0f;
        if (F16OUT) out16[(size_t)r * Nout + col] = (_Float16)x;
        else        out32[(size_t)r * Nout + col] = x;
      }
    }
  }
}

// ---------------------------------------------------------------------------
// Fused epilogue: out = relu(H @ WTu^T + bu) + relu(HO @ WTs^T + bs), f32 out.
__global__ __launch_bounds__(256) void final_fused_kernel(
    const _Float16* __restrict__ H,   const _Float16* __restrict__ WTu,
    const float* __restrict__ bu,
    const _Float16* __restrict__ HO,  const _Float16* __restrict__ WTs,
    const float* __restrict__ bs,
    float* __restrict__ out, int M) {
  const int wave = threadIdx.x >> 5;
  const int row0 = blockIdx.y * 64  + (wave & 3) * 16;
  const int col0 = blockIdx.x * 128 + (wave >> 2) * 64;

  v8f acc1[4] = {};
  for (int k0 = 0; k0 < LOWER; k0 += 32) {
    v16h a = load_a_frag(H, LOWER, row0, k0, M);
#pragma unroll
    for (int t = 0; t < 4; ++t) {
      v16h b = load_b_frag(WTu, LOWER, col0 + t * 16, k0);
      acc1[t] = __builtin_amdgcn_wmma_f32_16x16x32_f16(
          false, a, false, b, (short)0, acc1[t], false, false);
    }
  }
  v8f acc2[4] = {};
  for (int k0 = 0; k0 < HIDDEN; k0 += 32) {
    v16h a = load_a_frag(HO, HIDDEN, row0, k0, M);
#pragma unroll
    for (int t = 0; t < 4; ++t) {
      v16h b = load_b_frag(WTs, HIDDEN, col0 + t * 16, k0);
      acc2[t] = __builtin_amdgcn_wmma_f32_16x16x32_f16(
          false, a, false, b, (short)0, acc2[t], false, false);
    }
  }

  const int lane = threadIdx.x & 31;
  const int n  = lane & 15;
  const int mo = (lane >> 4) << 3;
#pragma unroll
  for (int t = 0; t < 4; ++t) {
    const int col = col0 + t * 16 + n;
    const float b1 = bu[col], b2 = bs[col];
#pragma unroll
    for (int v = 0; v < 8; ++v) {
      const int r = row0 + mo + v;
      if (r < M) {
        float x = acc1[t][v] + b1; x = x > 0.0f ? x : 0.0f;
        float y = acc2[t][v] + b2; y = y > 0.0f ? y : 0.0f;
        out[(size_t)r * HIDDEN + col] = x + y;
      }
    }
  }
}

// ---------------------------------------------------------------------------
// Prep: f32 [K][Nn] row-major -> f16 [Nn][K] row-major (batched over z).
__global__ void transpose_cvt_kernel(const float* __restrict__ src,
                                     _Float16* __restrict__ dst, int K, int Nn) {
  const size_t elems = (size_t)K * Nn;
  src += (size_t)blockIdx.z * elems;
  dst += (size_t)blockIdx.z * elems;
  const int idx = blockIdx.x * 256 + threadIdx.x;
  if (idx < (int)elems) {
    const int n = idx / K;
    const int k = idx - n * K;
    dst[idx] = (_Float16)src[(size_t)k * Nn + n];
  }
}

__global__ void cvt_f32_to_f16_kernel(const float* __restrict__ src,
                                      _Float16* __restrict__ dst, size_t n4) {
  const size_t i = (size_t)blockIdx.x * 256 + threadIdx.x;
  if (i < n4) {
    v4f s = *(const v4f*)(src + i * 4);
    v4h d;
    d[0] = (_Float16)s[0]; d[1] = (_Float16)s[1];
    d[2] = (_Float16)s[2]; d[3] = (_Float16)s[3];
    *(v4h*)(dst + i * 4) = d;
  }
}

__global__ void zero_kernel(float* __restrict__ p, size_t n4) {
  const size_t i = (size_t)blockIdx.x * 256 + threadIdx.x;
  if (i < n4) {
    v4f z = {};
    *(v4f*)(p + i * 4) = z;
  }
}

// ---------------------------------------------------------------------------
// Scatter: agg[dst] += t[rel, src]   (one wave per edge, 4 channels per lane,
// f32 hardware atomics -> global_atomic_add_f32)
__global__ __launch_bounds__(256) void scatter_kernel(
    const _Float16* __restrict__ T, const int* __restrict__ esrc,
    const int* __restrict__ edst, const int* __restrict__ etype,
    float* __restrict__ agg, int nEdges, int nNodes) {
  const int e = blockIdx.x * 8 + (threadIdx.x >> 5);
  if (e >= nEdges) return;
  const int lane = threadIdx.x & 31;
  const int src = esrc[e], dst = edst[e], rel = etype[e];
  const _Float16* trow = T + ((size_t)rel * nNodes + src) * LOWER + lane * 4;
  float* arow = agg + (size_t)dst * LOWER + lane * 4;
  v4h m = *(const v4h*)trow;
  unsafeAtomicAdd(arow + 0, (float)m[0]);
  unsafeAtomicAdd(arow + 1, (float)m[1]);
  unsafeAtomicAdd(arow + 2, (float)m[2]);
  unsafeAtomicAdd(arow + 3, (float)m[3]);
}

// h = (f16) relu(agg + bias)
__global__ void bias_relu_cvt_kernel(const float* __restrict__ agg,
                                     const float* __restrict__ bias,
                                     _Float16* __restrict__ h, int total) {
  const int idx = blockIdx.x * 256 + threadIdx.x;
  if (idx < total) {
    float x = agg[idx] + bias[idx & (LOWER - 1)];
    h[idx] = (_Float16)(x > 0.0f ? x : 0.0f);
  }
}

// ---------------------------------------------------------------------------
extern "C" void kernel_launch(void* const* d_in, const int* in_sizes, int n_in,
                              void* d_out, int out_size, void* d_ws, size_t ws_size,
                              hipStream_t stream) {
  const float* h_origin  = (const float*)d_in[0];
  const int*   edge_src  = (const int*)d_in[1];
  const int*   edge_dst  = (const int*)d_in[2];
  const int*   edge_type = (const int*)d_in[3];
  const float* W_lower   = (const float*)d_in[4];
  const float* b_lower   = (const float*)d_in[5];
  const float* W1        = (const float*)d_in[6];
  const float* b1        = (const float*)d_in[7];
  const float* W2        = (const float*)d_in[8];
  const float* b2        = (const float*)d_in[9];
  const float* W3        = (const float*)d_in[10];
  const float* b3        = (const float*)d_in[11];
  const float* W_upper   = (const float*)d_in[12];
  const float* b_upper   = (const float*)d_in[13];
  const float* W_skip    = (const float*)d_in[14];
  const float* b_skip    = (const float*)d_in[15];

  const int M = in_sizes[0] / HIDDEN;   // nodes
  const int E = in_sizes[1];            // edges

  // workspace carve (256B aligned)
  char* wsp = (char*)d_ws;
  auto carve = [&](size_t bytes) {
    char* p = wsp;
    wsp += (bytes + 255) & ~(size_t)255;
    return p;
  };
  _Float16* hO16 = (_Float16*)carve((size_t)M * HIDDEN * 2);
  _Float16* h    = (_Float16*)carve((size_t)M * LOWER * 2);
  _Float16* tbuf = (_Float16*)carve((size_t)NUM_RELS * M * LOWER * 2);
  float*    agg  = (float*)   carve((size_t)M * LOWER * 4);
  _Float16* WTlo = (_Float16*)carve((size_t)LOWER * HIDDEN * 2);
  _Float16* WT1  = (_Float16*)carve((size_t)NUM_RELS * LOWER * LOWER * 2);
  _Float16* WT2  = (_Float16*)carve((size_t)NUM_RELS * LOWER * LOWER * 2);
  _Float16* WT3  = (_Float16*)carve((size_t)NUM_RELS * LOWER * LOWER * 2);
  _Float16* WTup = (_Float16*)carve((size_t)HIDDEN * LOWER * 2);
  _Float16* WTsk = (_Float16*)carve((size_t)HIDDEN * HIDDEN * 2);

  // --- weight prep: transpose + f32->f16 (tiny; stays resident in L2) ---
  {
    const int tLoUp = HIDDEN * LOWER;        // 65536
    const int tRel  = LOWER * LOWER;         // 16384
    const int tSkip = HIDDEN * HIDDEN;       // 262144
    transpose_cvt_kernel<<<dim3((tLoUp + 255) / 256, 1, 1), 256, 0, stream>>>(
        W_lower, WTlo, HIDDEN, LOWER);
    transpose_cvt_kernel<<<dim3((tRel + 255) / 256, 1, NUM_RELS), 256, 0, stream>>>(
        W1, WT1, LOWER, LOWER);
    transpose_cvt_kernel<<<dim3((tRel + 255) / 256, 1, NUM_RELS), 256, 0, stream>>>(
        W2, WT2, LOWER, LOWER);
    transpose_cvt_kernel<<<dim3((tRel + 255) / 256, 1, NUM_RELS), 256, 0, stream>>>(
        W3, WT3, LOWER, LOWER);
    transpose_cvt_kernel<<<dim3((tLoUp + 255) / 256, 1, 1), 256, 0, stream>>>(
        W_upper, WTup, LOWER, HIDDEN);
    transpose_cvt_kernel<<<dim3((tSkip + 255) / 256, 1, 1), 256, 0, stream>>>(
        W_skip, WTsk, HIDDEN, HIDDEN);
  }
  // h_origin -> f16 (feeds both lower GEMM and the skip GEMM)
  {
    const size_t n4 = (size_t)M * HIDDEN / 4;
    cvt_f32_to_f16_kernel<<<(unsigned)((n4 + 255) / 256), 256, 0, stream>>>(
        h_origin, hO16, n4);
  }

  const dim3 blk(256);
  const int gy = (M + 63) / 64;

  // lower: h = relu(hO16 @ W_lower + b_lower), f16 out
  wmma_gemm_kernel<true, true, true><<<dim3(1, gy, 1), blk, 0, stream>>>(
      hO16, WTlo, b_lower, h, M, HIDDEN, LOWER, 0L, 0L);

  const float* lbias[3] = {b1, b2, b3};
  _Float16* lwt[3] = {WT1, WT2, WT3};
  const size_t n4agg = (size_t)M * LOWER / 4;

  for (int l = 0; l < 3; ++l) {
    zero_kernel<<<(unsigned)((n4agg + 255) / 256), 256, 0, stream>>>(agg, n4agg);
    // t[r] = h @ W_l[r]  (batched over relations; A shared -> L2 hit)
    wmma_gemm_kernel<false, true, false><<<dim3(1, gy, NUM_RELS), blk, 0, stream>>>(
        h, lwt[l], nullptr, tbuf, M, LOWER, LOWER,
        (long)LOWER * LOWER, (long)M * LOWER);
    // agg[dst] += t[rel, src]
    scatter_kernel<<<(E + 7) / 8, 256, 0, stream>>>(
        tbuf, edge_src, edge_dst, edge_type, agg, E, M);
    // h = relu(agg + b)
    bias_relu_cvt_kernel<<<((M * LOWER) + 255) / 256, 256, 0, stream>>>(
        agg, lbias[l], h, M * LOWER);
  }

  // out = relu(h @ W_upper + b_upper) + relu(h_origin @ W_skip + b_skip)
  final_fused_kernel<<<dim3(HIDDEN / 128, gy, 1), blk, 0, stream>>>(
      h, WTup, b_upper, hO16, WTsk, b_skip, (float*)d_out, M);
}
